// ConditionInjectionBlock_39298950758616
// MI455X (gfx1250) — matrix operand order repaired and checked by reading, not compiled, not run on hardware
//
#include <hip/hip_runtime.h>

typedef __bf16 bf16;
typedef __attribute__((ext_vector_type(16))) __bf16 v16bf;
typedef __attribute__((ext_vector_type(8)))  __bf16 v8bf;
typedef __attribute__((ext_vector_type(8)))  float  v8f;
typedef unsigned int uint32;

#define WSZ      8
#define HEADS    8
#define DIM      256
#define HD       32
#define WS2      64
#define NQ       256                   // queries per window (M*ws2)
#define NWIN     1024                  // B * nW
#define QROWS    (NWIN * NQ)           // 262144
#define KVROWS   (NWIN * WS2)          // 65536
#define ATT_SCALE 0.17677669529663687f // 1/sqrt(32)

#define WPITCH 270   // bf16 elems per row of transposed weight in LDS (135 dwords, coprime with 64 banks)
#define BPITCH 68    // f32 elems per row of bias tile
#define PPITCH 70    // bf16 elems per row of P tile (35 dwords, coprime with 64 banks)

union BF16Frag { v16bf v; v8bf v8[2]; __bf16 h[16]; uint32 u[8]; };
union F32Frag  { v8f v; float f[8]; };

static __device__ __forceinline__ v8f wmma_bf16(v16bf a, v16bf b, v8f c) {
  // D = A(16x32 bf16) * B(32x16 bf16) + C(16x16 f32)
  return __builtin_amdgcn_wmma_f32_16x16x32_bf16(false, a, false, b, (short)0, c, false, false);
}

// Wave-local LDS ordering: same-wave DS ops complete in order; this is a
// compiler barrier + wait so cross-lane LDS store->load within one wave is
// safe without a workgroup barrier (all attn LDS regions are wave-private).
static __device__ __forceinline__ void lds_wave_fence() {
  asm volatile("s_wait_dscnt 0" ::: "memory");
}

// ---------------------------------------------------------------------------
// Kernel 1: Q = (g @ Wq + bq) * scale   -> bf16 [QROWS][DIM]
// 256 threads = 8 waves; each wave: 16 rows x 256 cols (16 WMMA accumulators).
// ---------------------------------------------------------------------------
__global__ __launch_bounds__(256)
void qproj_kernel(const float* __restrict__ g, const float* __restrict__ Wq,
                  const float* __restrict__ bq, bf16* __restrict__ q_ws) {
  __shared__ bf16 WT[DIM * WPITCH];          // Wq^T, bf16
  const int t = threadIdx.x;
  for (int k = 0; k < DIM; ++k)
    WT[t * WPITCH + k] = (bf16)Wq[k * DIM + t];
  __syncthreads();

  const int lane = t & 31, wv = t >> 5;
  const int l16 = lane & 15, lh = lane >> 4;
  const int rowBase = blockIdx.x * 128 + wv * 16;

  float bqv[16];
#pragma unroll
  for (int nt = 0; nt < 16; ++nt) bqv[nt] = bq[nt * 16 + l16];

  const float* arow = g + (size_t)(rowBase + l16) * DIM;
  v8f acc[16] = {};
  const uint32* wb = (const uint32*)WT;

  for (int kc = 0; kc < 8; ++kc) {
    BF16Frag a;
    const float4* p0 = (const float4*)(arow + kc * 32 + lh * 16);
    const float4* p1 = (const float4*)(arow + kc * 32 + 8 + lh * 16);
    float4 f0 = p0[0], f1 = p0[1], f2 = p1[0], f3 = p1[1];
    a.h[0]=(bf16)f0.x; a.h[1]=(bf16)f0.y; a.h[2]=(bf16)f0.z; a.h[3]=(bf16)f0.w;
    a.h[4]=(bf16)f1.x; a.h[5]=(bf16)f1.y; a.h[6]=(bf16)f1.z; a.h[7]=(bf16)f1.w;
    a.h[8]=(bf16)f2.x; a.h[9]=(bf16)f2.y; a.h[10]=(bf16)f2.z; a.h[11]=(bf16)f2.w;
    a.h[12]=(bf16)f3.x; a.h[13]=(bf16)f3.y; a.h[14]=(bf16)f3.z; a.h[15]=(bf16)f3.w;
#pragma unroll
    for (int nt = 0; nt < 16; ++nt) {
      BF16Frag b;
      int base = ((nt * 16 + l16) * WPITCH + kc * 32 + lh * 16) >> 1;
#pragma unroll
      for (int i = 0; i < 8; ++i) b.u[i] = wb[base + i];
      acc[nt] = wmma_bf16(a.v, b.v, acc[nt]);
    }
  }

  const int rowC = rowBase + lh * 8;
#pragma unroll
  for (int nt = 0; nt < 16; ++nt) {
    F32Frag c; c.v = acc[nt];
#pragma unroll
    for (int r = 0; r < 8; ++r)
      q_ws[(size_t)(rowC + r) * DIM + nt * 16 + l16] =
          (bf16)((c.f[r] + bqv[nt]) * ATT_SCALE);
  }
}

// ---------------------------------------------------------------------------
// Kernel 2: KV projection. blockIdx.y: 0 -> K half, 1 -> V half.
// K stored [win][key][ch]; V stored transposed [win][ch][key].
// ---------------------------------------------------------------------------
__global__ __launch_bounds__(256)
void kvproj_kernel(const float* __restrict__ img, const float* __restrict__ Wkv,
                   const float* __restrict__ bkv, bf16* __restrict__ k_ws,
                   bf16* __restrict__ v_ws) {
  const int half = blockIdx.y;
  __shared__ bf16 WT[DIM * WPITCH];
  const int t = threadIdx.x;
  for (int k = 0; k < DIM; ++k)
    WT[t * WPITCH + k] = (bf16)Wkv[k * 2 * DIM + half * DIM + t];
  __syncthreads();

  const int lane = t & 31, wv = t >> 5;
  const int l16 = lane & 15, lh = lane >> 4;
  const int rowBase = blockIdx.x * 128 + wv * 16;

  float bv[16];
#pragma unroll
  for (int nt = 0; nt < 16; ++nt) bv[nt] = bkv[half * DIM + nt * 16 + l16];

  // window-partitioned source row for this lane
  const int rowA = rowBase + l16;
  const int bwA = rowA >> 6, eA = rowA & 63;
  const int bb = bwA >> 8, wxy = bwA & 255;
  const int wy = wxy >> 4, wx = wxy & 15;
  const int rr = eA >> 3, cc = eA & 7;
  const float* arow = img + (size_t)(((bb * 128 + wy * 8 + rr) * 128) + wx * 8 + cc) * DIM;

  v8f acc[16] = {};
  const uint32* wb = (const uint32*)WT;

  for (int kc = 0; kc < 8; ++kc) {
    BF16Frag a;
    const float4* p0 = (const float4*)(arow + kc * 32 + lh * 16);
    const float4* p1 = (const float4*)(arow + kc * 32 + 8 + lh * 16);
    float4 f0 = p0[0], f1 = p0[1], f2 = p1[0], f3 = p1[1];
    a.h[0]=(bf16)f0.x; a.h[1]=(bf16)f0.y; a.h[2]=(bf16)f0.z; a.h[3]=(bf16)f0.w;
    a.h[4]=(bf16)f1.x; a.h[5]=(bf16)f1.y; a.h[6]=(bf16)f1.z; a.h[7]=(bf16)f1.w;
    a.h[8]=(bf16)f2.x; a.h[9]=(bf16)f2.y; a.h[10]=(bf16)f2.z; a.h[11]=(bf16)f2.w;
    a.h[12]=(bf16)f3.x; a.h[13]=(bf16)f3.y; a.h[14]=(bf16)f3.z; a.h[15]=(bf16)f3.w;
#pragma unroll
    for (int nt = 0; nt < 16; ++nt) {
      BF16Frag b;
      int base = ((nt * 16 + l16) * WPITCH + kc * 32 + lh * 16) >> 1;
#pragma unroll
      for (int i = 0; i < 8; ++i) b.u[i] = wb[base + i];
      acc[nt] = wmma_bf16(a.v, b.v, acc[nt]);
    }
  }

  const int rowC = rowBase + lh * 8;
#pragma unroll
  for (int nt = 0; nt < 16; ++nt) {
    F32Frag c; c.v = acc[nt];
#pragma unroll
    for (int r = 0; r < 8; ++r) {
      float o = c.f[r] + bv[nt];
      int row = rowC + r;
      int bw2 = row >> 6, key = row & 63;
      int ch = nt * 16 + l16;
      if (half == 0)
        k_ws[((size_t)bw2 * 64 + key) * DIM + ch] = (bf16)o;
      else
        v_ws[((size_t)bw2 * DIM + ch) * 64 + key] = (bf16)o;
    }
  }
}

// ---------------------------------------------------------------------------
// Kernel 3: attention. 1 block per window, 1 wave per head.
// K/V fragments register-resident; softmax in registers via shfl_xor;
// P re-laid-out via a small wave-private LDS tile (C-frag -> A-frag).
// No workgroup barriers: every LDS region is private to one wave, and
// same-wave DS ops are processed in order (ordered via s_wait_dscnt fences).
// ---------------------------------------------------------------------------
__global__ __launch_bounds__(256)
void attn_kernel(const bf16* __restrict__ q_ws, const bf16* __restrict__ k_ws,
                 const bf16* __restrict__ v_ws, const float* __restrict__ bias_table,
                 bf16* __restrict__ attn_ws) {
  __shared__ float biasT[HEADS * WS2 * BPITCH];   // 139264 B
  __shared__ bf16  Pt[HEADS * 16 * PPITCH];       // 17920 B
  const int t = threadIdx.x, lane = t & 31, h = t >> 5;
  const int l16 = lane & 15, lh = lane >> 4;
  const int bw = blockIdx.x;

  // materialize this head's relative-position bias tile [64 q][64 k]
  for (int i = lane; i < WS2 * WS2; i += 32) {
    int q64 = i >> 6, kk = i & 63;
    int qi = q64 >> 3, qj = q64 & 7, ki = kk >> 3, kj = kk & 7;
    int idx = (qi - ki + 7) * 15 + (qj - kj + 7);
    biasT[h * WS2 * BPITCH + q64 * BPITCH + kk] = bias_table[idx * HEADS + h];
  }

  // K fragments (B-operand of Q@K^T): 4 key tiles
  BF16Frag kf[4];
#pragma unroll
  for (int kt = 0; kt < 4; ++kt) {
    int key = kt * 16 + l16;
    const v8bf* p = (const v8bf*)(k_ws + ((size_t)bw * 64 + key) * DIM + h * HD + lh * 16);
    kf[kt].v8[0] = p[0]; kf[kt].v8[1] = p[1];
  }
  // V fragments (B-operand of P@V): [kchunk][ntile]
  BF16Frag vf[2][2];
#pragma unroll
  for (int kc = 0; kc < 2; ++kc)
#pragma unroll
    for (int nt = 0; nt < 2; ++nt) {
      int d = nt * 16 + l16;
      int key0 = kc * 32 + lh * 16;
      const v8bf* p = (const v8bf*)(v_ws + ((size_t)bw * DIM + h * HD + d) * 64 + key0);
      vf[kc][nt].v8[0] = p[0]; vf[kc][nt].v8[1] = p[1];
    }

  bf16* Pw = Pt + h * 16 * PPITCH;
  const float* bh = biasT + h * WS2 * BPITCH;
  const uint32* pl = (const uint32*)Pw;
  lds_wave_fence();  // bias tile visible to this wave's cross-lane reads

  for (int qt = 0; qt < 16; ++qt) {
    // Q A-fragment (K-dim = head_dim = 32, exactly one chunk)
    BF16Frag qa;
    {
      const bf16* qr = q_ws + ((size_t)bw * NQ + qt * 16 + l16) * DIM + h * HD + lh * 16;
      qa.v8[0] = *(const v8bf*)qr;
      qa.v8[1] = *(const v8bf*)(qr + 8);
    }
    // S = Q*scale @ K^T  (4 key tiles)
    F32Frag s[4];
#pragma unroll
    for (int kt = 0; kt < 4; ++kt) {
      v8f z = {};
      s[kt].v = wmma_bf16(qa.v, kf[kt].v, z);
    }
    // + relative position bias
    const int qrel = (qt & 3) * 16 + lh * 8;
#pragma unroll
    for (int kt = 0; kt < 4; ++kt)
#pragma unroll
      for (int r = 0; r < 8; ++r)
        s[kt].f[r] += bh[(qrel + r) * BPITCH + kt * 16 + l16];
    // row softmax over 64 keys: each row lives in 16 lanes x 4 fragments
#pragma unroll
    for (int r = 0; r < 8; ++r) {
      float mx = fmaxf(fmaxf(s[0].f[r], s[1].f[r]), fmaxf(s[2].f[r], s[3].f[r]));
      mx = fmaxf(mx, __shfl_xor(mx, 1, 32));
      mx = fmaxf(mx, __shfl_xor(mx, 2, 32));
      mx = fmaxf(mx, __shfl_xor(mx, 4, 32));
      mx = fmaxf(mx, __shfl_xor(mx, 8, 32));
      float e0 = __expf(s[0].f[r] - mx);
      float e1 = __expf(s[1].f[r] - mx);
      float e2 = __expf(s[2].f[r] - mx);
      float e3 = __expf(s[3].f[r] - mx);
      float sum = e0 + e1 + e2 + e3;
      sum += __shfl_xor(sum, 1, 32);
      sum += __shfl_xor(sum, 2, 32);
      sum += __shfl_xor(sum, 4, 32);
      sum += __shfl_xor(sum, 8, 32);
      float inv = __builtin_amdgcn_rcpf(sum);
      int m = lh * 8 + r;
      Pw[m * PPITCH +  0 + l16] = (bf16)(e0 * inv);
      Pw[m * PPITCH + 16 + l16] = (bf16)(e1 * inv);
      Pw[m * PPITCH + 32 + l16] = (bf16)(e2 * inv);
      Pw[m * PPITCH + 48 + l16] = (bf16)(e3 * inv);
    }
    lds_wave_fence();  // P stores ordered before cross-lane P loads (same wave)
    // P A-fragments from LDS (2 K-chunks of 32)
    BF16Frag pa[2];
#pragma unroll
    for (int c = 0; c < 2; ++c) {
      int k0 = c * 32 + lh * 16;
      int base0 = (l16 * PPITCH + k0) >> 1;
      int base1 = (l16 * PPITCH + k0 + 8) >> 1;
#pragma unroll
      for (int i = 0; i < 4; ++i) pa[c].u[i] = pl[base0 + i];
#pragma unroll
      for (int i = 0; i < 4; ++i) pa[c].u[4 + i] = pl[base1 + i];
    }
    lds_wave_fence();  // P loads ordered before next iteration's overwrites
    // out = P @ V  (2 d-tiles)
#pragma unroll
    for (int nt = 0; nt < 2; ++nt) {
      v8f o = {};
      o = wmma_bf16(pa[0].v, vf[0][nt].v, o);
      o = wmma_bf16(pa[1].v, vf[1][nt].v, o);
      F32Frag of; of.v = o;
#pragma unroll
      for (int r = 0; r < 8; ++r) {
        int row = qt * 16 + lh * 8 + r;
        attn_ws[((size_t)bw * NQ + row) * DIM + h * HD + nt * 16 + l16] = (bf16)of.f[r];
      }
    }
  }
}

// ---------------------------------------------------------------------------
// Kernel 4: out = attn @ Wp + bp  (bf16 in, f32 out)
// ---------------------------------------------------------------------------
__global__ __launch_bounds__(256)
void proj_kernel(const bf16* __restrict__ a_in, const float* __restrict__ Wp,
                 const float* __restrict__ bp, float* __restrict__ out) {
  __shared__ bf16 WT[DIM * WPITCH];
  const int t = threadIdx.x;
  for (int k = 0; k < DIM; ++k)
    WT[t * WPITCH + k] = (bf16)Wp[k * DIM + t];
  __syncthreads();

  const int lane = t & 31, wv = t >> 5;
  const int l16 = lane & 15, lh = lane >> 4;
  const int rowBase = blockIdx.x * 128 + wv * 16;

  float bpv[16];
#pragma unroll
  for (int nt = 0; nt < 16; ++nt) bpv[nt] = bp[nt * 16 + l16];

  const bf16* arow = a_in + (size_t)(rowBase + l16) * DIM;
  v8f acc[16] = {};
  const uint32* wb = (const uint32*)WT;

  for (int kc = 0; kc < 8; ++kc) {
    BF16Frag a;
    a.v8[0] = *(const v8bf*)(arow + kc * 32 + lh * 16);
    a.v8[1] = *(const v8bf*)(arow + kc * 32 + 8 + lh * 16);
#pragma unroll
    for (int nt = 0; nt < 16; ++nt) {
      BF16Frag b;
      int base = ((nt * 16 + l16) * WPITCH + kc * 32 + lh * 16) >> 1;
#pragma unroll
      for (int i = 0; i < 8; ++i) b.u[i] = wb[base + i];
      acc[nt] = wmma_bf16(a.v, b.v, acc[nt]);
    }
  }

  const int rowC = rowBase + lh * 8;
#pragma unroll
  for (int nt = 0; nt < 16; ++nt) {
    F32Frag c; c.v = acc[nt];
#pragma unroll
    for (int r = 0; r < 8; ++r)
      out[(size_t)(rowC + r) * DIM + nt * 16 + l16] = c.f[r] + bpv[nt];
  }
}

// ---------------------------------------------------------------------------
extern "C" void kernel_launch(void* const* d_in, const int* in_sizes, int n_in,
                              void* d_out, int out_size, void* d_ws, size_t ws_size,
                              hipStream_t stream) {
  const float* g    = (const float*)d_in[0];  // [4, 65536, 256]
  const float* img  = (const float*)d_in[1];  // [4, 128, 128, 256]
  const float* bias = (const float*)d_in[2];  // [225, 8]
  const float* Wq   = (const float*)d_in[3];
  const float* bq   = (const float*)d_in[4];
  const float* Wkv  = (const float*)d_in[5];
  const float* bkv  = (const float*)d_in[6];
  const float* Wp   = (const float*)d_in[7];
  const float* bp   = (const float*)d_in[8];
  float* out = (float*)d_out;

  const size_t QBYTES = (size_t)QROWS * DIM * sizeof(bf16);   // 134 MB
  const size_t KBYTES = (size_t)KVROWS * DIM * sizeof(bf16);  // 33.5 MB
  char* base = (char*)d_ws;
  bf16* q_ws    = (bf16*)base;
  bf16* k_ws    = (bf16*)(base + QBYTES);
  bf16* v_ws    = (bf16*)(base + QBYTES + KBYTES);
  bf16* attn_ws = (bf16*)(base + QBYTES + 2 * KBYTES);

  qproj_kernel<<<QROWS / 128, 256, 0, stream>>>(g, Wq, bq, q_ws);
  kvproj_kernel<<<dim3(KVROWS / 128, 2), 256, 0, stream>>>(img, Wkv, bkv, k_ws, v_ws);
  attn_kernel<<<NWIN, 256, 0, stream>>>(q_ws, k_ws, v_ws, bias, attn_ws);
  proj_kernel<<<QROWS / 128, 256, 0, stream>>>(attn_ws, Wp, bp, out);
}
